// QuantizedLinear_49246095015985
// MI455X (gfx1250) — 8-bit matrix operand order repaired and checked by reading, compiled, not run on hardware
//
#include <hip/hip_runtime.h>

typedef __attribute__((ext_vector_type(8))) int v8i;
typedef __attribute__((ext_vector_type(4))) int v4i;

constexpr int Mtot = 4 * 2048;
constexpr int Ktot = 4096;
constexpr int Ntot = 4096;
constexpr int KB   = Ktot / 64;

__device__ __forceinline__ int a_dword_index(int mt, int ml, int k) {
    int kt = k >> 6, kl = k & 63;
    int v    = ((kl >> 5) << 2) | ((kl & 15) >> 2);
    int lane = (((kl >> 4) & 1) << 4) | ml;
    return (mt * KB + kt) * 256 + lane * 8 + v;
}
__device__ __forceinline__ int b_dword_index(int nt, int nl, int k) {
    int kt = k >> 6, kl = k & 63;
    int v    = ((kl >> 5) << 2) | ((kl & 15) >> 2);
    int lane = (((kl >> 4) & 1) << 4) | nl;
    return (nt * KB + kt) * 256 + lane * 8 + v;
}

__global__ void __launch_bounds__(256) quantize_x_kernel(
    const float* __restrict__ x,
    const float* __restrict__ act_scale,
    const int*   __restrict__ act_zp,
    unsigned int* __restrict__ xq_blk,
    int* __restrict__ xsum)
{
    __shared__ int red[256];
    const int m  = blockIdx.x;
    const int t  = threadIdx.x;
    const float inv_s = 1.0f / act_scale[0];
    const float zp    = (float)act_zp[0];
    const int mt = m >> 4, ml = m & 15;

    const float4* xr = (const float4*)(x + (size_t)m * Ktot);
    int psum = 0;
#pragma unroll
    for (int j = 0; j < 4; ++j) {
        const int kq = t * 4 + j;
        const int k  = kq * 4;
        float4 f = xr[kq];
        int q0 = (int)fminf(fmaxf(rintf(f.x * inv_s) + zp, 0.0f), 255.0f);
        int q1 = (int)fminf(fmaxf(rintf(f.y * inv_s) + zp, 0.0f), 255.0f);
        int q2 = (int)fminf(fmaxf(rintf(f.z * inv_s) + zp, 0.0f), 255.0f);
        int q3 = (int)fminf(fmaxf(rintf(f.w * inv_s) + zp, 0.0f), 255.0f);
        psum += q0 + q1 + q2 + q3;
        unsigned int packed = (unsigned)q0 | ((unsigned)q1 << 8) |
                              ((unsigned)q2 << 16) | ((unsigned)q3 << 24);
        xq_blk[a_dword_index(mt, ml, k)] = packed;
    }
    red[t] = psum;
    __syncthreads();
#pragma unroll
    for (int s = 128; s > 0; s >>= 1) {
        if (t < s) red[t] += red[t + s];
        __syncthreads();
    }
    if (t == 0) xsum[m] = red[0];
}

__global__ void __launch_bounds__(256) pack_w_kernel(
    const int* __restrict__ weight_q,
    const int* __restrict__ wzp,
    const int* __restrict__ act_zp,
    unsigned int* __restrict__ wq_blk,
    int* __restrict__ woff)
{
    __shared__ int red[256];
    const int n  = blockIdx.x;
    const int t  = threadIdx.x;
    const int nt = n >> 4, nl = n & 15;

    const int4* wr = (const int4*)(weight_q + (size_t)n * Ktot);
    int psum = 0;
#pragma unroll
    for (int j = 0; j < 4; ++j) {
        const int kq = t * 4 + j;
        const int k  = kq * 4;
        int4 w = wr[kq];
        psum += w.x + w.y + w.z + w.w;
        unsigned int packed = (unsigned)(w.x & 255) | ((unsigned)(w.y & 255) << 8) |
                              ((unsigned)(w.z & 255) << 16) | ((unsigned)(w.w & 255) << 24);
        wq_blk[b_dword_index(nt, nl, k)] = packed;
    }
    red[t] = psum;
    __syncthreads();
#pragma unroll
    for (int s = 128; s > 0; s >>= 1) {
        if (t < s) red[t] += red[t + s];
        __syncthreads();
    }
    if (t == 0) woff[n] = act_zp[0] * (Ktot * wzp[n] - red[0]);
}

__global__ void __launch_bounds__(256) gemm_iu8_kernel(
    const v4i* __restrict__ aq,
    const v4i* __restrict__ bq,
    const int*   __restrict__ xsum,
    const int*   __restrict__ woff,
    const int*   __restrict__ wzp,
    const float* __restrict__ wscale,
    const float* __restrict__ bias,
    const float* __restrict__ act_scale,
    float* __restrict__ out)
{
    const int lane = threadIdx.x & 31;
    const int w    = threadIdx.x >> 5;
    const int wm   = w >> 2, wn = w & 3;
    const int m0   = blockIdx.y * 128 + wm * 64;
    const int n0   = blockIdx.x * 256 + wn * 64;
    const int mt0  = m0 >> 4, nt0 = n0 >> 4;

    const v4i* ap[4];
    const v4i* bp[4];
#pragma unroll
    for (int i = 0; i < 4; ++i)
        ap[i] = aq + ((size_t)(mt0 + i) * KB) * 64 + lane * 2;
#pragma unroll
    for (int j = 0; j < 4; ++j)
        bp[j] = bq + ((size_t)(nt0 + j) * KB) * 64 + lane * 2;

    v8i acc[4][4];
#pragma unroll
    for (int i = 0; i < 4; ++i)
#pragma unroll
        for (int j = 0; j < 4; ++j)
            acc[i][j] = (v8i){0, 0, 0, 0, 0, 0, 0, 0};

    union Frag { v8i v; v4i q[2]; };

#define LOAD_PHASE(AF, BF, KT)                                              \
    do {                                                                    \
        const int _off = (KT) * 64;                                         \
        _Pragma("unroll")                                                   \
        for (int _i = 0; _i < 4; ++_i) {                                    \
            Frag _u;                                                        \
            _u.q[0] = ap[_i][_off];                                         \
            _u.q[1] = ap[_i][_off + 1];                                     \
            (AF)[_i] = _u.v;                                                \
            __builtin_prefetch(&ap[_i][_off + 128], 0, 3);                  \
        }                                                                   \
        _Pragma("unroll")                                                   \
        for (int _j = 0; _j < 4; ++_j) {                                    \
            Frag _u;                                                        \
            _u.q[0] = bp[_j][_off];                                         \
            _u.q[1] = bp[_j][_off + 1];                                     \
            (BF)[_j] = _u.v;                                                \
            __builtin_prefetch(&bp[_j][_off + 128], 0, 3);                  \
        }                                                                   \
    } while (0)

#define WMMA_PHASE(AF, BF)                                                  \
    do {                                                                    \
        _Pragma("unroll")                                                   \
        for (int _j = 0; _j < 4; ++_j) {                                    \
            _Pragma("unroll")                                               \
            for (int _i = 0; _i < 4; ++_i)                                  \
                acc[_i][_j] = __builtin_amdgcn_wmma_i32_16x16x64_iu8(       \
                    false, (AF)[_i], false, (BF)[_j], acc[_i][_j],          \
                    false, false);                                          \
        }                                                                   \
    } while (0)

    v8i aA[4], bA[4], aB[4], bB[4];
    LOAD_PHASE(aA, bA, 0);

    int kt = 0;
    for (; kt < KB - 2; kt += 2) {
        LOAD_PHASE(aB, bB, kt + 1);
        WMMA_PHASE(aA, bA);
        LOAD_PHASE(aA, bA, kt + 2);
        WMMA_PHASE(aB, bB);
    }
    LOAD_PHASE(aB, bB, KB - 1);
    WMMA_PHASE(aA, bA);
    WMMA_PHASE(aB, bB);

#undef LOAD_PHASE
#undef WMMA_PHASE

    const float s_a = act_scale[0];
    const int   nl  = lane & 15;
    const int   hh  = (lane >> 4) << 3;
#pragma unroll
    for (int i = 0; i < 4; ++i) {
        const int mbase = m0 + i * 16 + hh;
        int xs[8];
#pragma unroll
        for (int r = 0; r < 8; ++r) xs[r] = xsum[mbase + r];
#pragma unroll
        for (int j = 0; j < 4; ++j) {
            const int   n   = n0 + j * 16 + nl;
            const float sc  = s_a * wscale[n];
            const int   zn  = wzp[n];
            const int   off = woff[n];
            const float bn  = bias[n];
#pragma unroll
            for (int r = 0; r < 8; ++r) {
                const int corrected = acc[i][j][r] - zn * xs[r] + off;
                out[(size_t)(mbase + r) * Ntot + n] = sc * (float)corrected + bn;
            }
        }
    }
}

extern "C" void kernel_launch(void* const* d_in, const int* in_sizes, int n_in,
                              void* d_out, int out_size, void* d_ws, size_t ws_size,
                              hipStream_t stream) {
    const float* x    = (const float*)d_in[0];
    const int*   wq   = (const int*)d_in[1];
    const float* wsc  = (const float*)d_in[2];
    const int*   wzp  = (const int*)d_in[3];
    const float* asc  = (const float*)d_in[4];
    const int*   azp  = (const int*)d_in[5];
    const float* bias = (const float*)d_in[6];
    float*       out  = (float*)d_out;

    char* ws = (char*)d_ws;
    unsigned int* xq_blk = (unsigned int*)ws;
    unsigned int* wq_blk = (unsigned int*)(ws + (size_t)Mtot * Ktot);
    int* xsum = (int*)(ws + (size_t)Mtot * Ktot + (size_t)Ntot * Ktot);
    int* woff = xsum + Mtot;

    quantize_x_kernel<<<Mtot, 256, 0, stream>>>(x, asc, azp, xq_blk, xsum);
    pack_w_kernel<<<Ntot, 256, 0, stream>>>(wq, wzp, azp, wq_blk, woff);

    dim3 grid(Ntot / 256, Mtot / 128);
    gemm_iu8_kernel<<<grid, 256, 0, stream>>>(
        (const v4i*)xq_blk, (const v4i*)wq_blk,
        xsum, woff, wzp, wsc, bias, asc, out);
}
